// MetapathAggregation_17248588660756
// MI455X (gfx1250) — matrix operand order repaired and checked by reading, not compiled, not run on hardware
//
#include <hip/hip_runtime.h>
#include <hip/hip_bf16.h>

typedef __attribute__((ext_vector_type(16))) _Float16 v16h;
typedef __attribute__((ext_vector_type(8)))  _Float16 v8h;
typedef __attribute__((ext_vector_type(8)))  float    v8f;

#define NA 30000
#define NB 30000
#define VV 2
#define DD 128
#define HEADS 4
#define NEDGE 480000

__device__ __forceinline__ float waveReduceSum(float v) {
    v += __shfl_xor(v, 1, 32);
    v += __shfl_xor(v, 2, 32);
    v += __shfl_xor(v, 4, 32);
    v += __shfl_xor(v, 8, 32);
    v += __shfl_xor(v, 16, 32);
    return v;
}

// ---------------- zero fill (float4 granularity) ----------------
__global__ void k_zero(float* __restrict__ p, int n4) {
    int i = blockIdx.x * blockDim.x + threadIdx.x;
    if (i < n4) ((float4*)p)[i] = make_float4(0.f, 0.f, 0.f, 0.f);
}

// ---------------- SpMM-T: out[dst] += feat[src] * val ----------------
__global__ void k_spmm(const float* __restrict__ feat,
                       const int* __restrict__ src, const int* __restrict__ dst,
                       const float* __restrict__ val,
                       float* __restrict__ out, int nEdges) {
    int tid = blockIdx.x * blockDim.x + threadIdx.x;
    int e = tid >> 6;      // 64 float4-chunks per edge (V*D = 256 floats)
    int c = tid & 63;
    if (e >= nEdges) return;
    int s = src[e], d = dst[e];
    float v = val[e];
    float4 f = ((const float4*)(feat + (size_t)s * 256))[c];
    float* o = out + (size_t)d * 256 + c * 4;
    unsafeAtomicAdd(o + 0, f.x * v);
    unsafeAtomicAdd(o + 1, f.y * v);
    unsafeAtomicAdd(o + 2, f.z * v);
    unsafeAtomicAdd(o + 3, f.w * v);
}

// ---------------- row-wise L2 normalize (wave per 128-float row) ----------------
__global__ void k_l2norm(float* __restrict__ x, int rows) {
    int wave = (blockIdx.x * blockDim.x + threadIdx.x) >> 5;
    int lane = threadIdx.x & 31;
    if (wave >= rows) return;
    float* r = x + (size_t)wave * 128;
    float4 t = ((float4*)r)[lane];
    float ss = waveReduceSum(t.x * t.x + t.y * t.y + t.z * t.z + t.w * t.w);
    float sc = 1.0f / fmaxf(sqrtf(ss), 1e-12f);
    t.x *= sc; t.y *= sc; t.z *= sc; t.w *= sc;
    ((float4*)r)[lane] = t;
}

// ---------------- f32 -> f16 row convert (wave per 128-col row; strided input) -----
__global__ void k_cvt_f16(const float* __restrict__ in, int inStride,
                          _Float16* __restrict__ out, int rows) {
    int wave = (blockIdx.x * blockDim.x + threadIdx.x) >> 5;
    int lane = threadIdx.x & 31;
    if (wave >= rows) return;
    float4 t = ((const float4*)(in + (size_t)wave * inStride))[lane];
    typedef __attribute__((ext_vector_type(4))) _Float16 v4h;
    v4h h = { (_Float16)t.x, (_Float16)t.y, (_Float16)t.z, (_Float16)t.w };
    *(v4h*)(out + (size_t)wave * 128 + lane * 4) = h;
}

// ---------------- WMMA GEMM:  C[M,N] = A[M,128] * W[N,128]^T + bias ----------------
// f16 operands, f32 accumulate. One wave computes a 16x64 strip (4 N-tiles),
// re-using the A fragment across 4 WMMAs per k-step. Pure b128 loads + WMMA.
__global__ void k_gemm_wmma(const _Float16* __restrict__ A,   // [M,128] f16
                            const _Float16* __restrict__ W,   // [N,128] f16
                            const float* __restrict__ bias,   // [N] f32 or null
                            float* __restrict__ C, int ldc,
                            int Mtiles, int Ngroups) {        // Ngroup = 4 N-tiles
    int gw   = (blockIdx.x * blockDim.x + threadIdx.x) >> 5;
    int lane = threadIdx.x & 31;
    if (gw >= Mtiles * Ngroups) return;          // wave-uniform: EXEC stays all-ones
    int ng = gw % Ngroups;
    int mt = gw / Ngroups;
    int half = lane >> 4;
    int rl   = lane & 15;

    const _Float16* Arow  = A + (size_t)(mt * 16 + rl) * 128;
    const _Float16* Wrow0 = W + (size_t)(ng * 64 + rl) * 128;

    v8f acc[4] = {};
#pragma unroll
    for (int ks = 0; ks < 4; ++ks) {
        int kb = ks * 32;
        // A 16x32 f16 fragment: per-lane halves at K = kb+half*8..+7 and kb+16+half*8..+7
        union { v16h v; v8h h[2]; } af;
        af.h[0] = *(const v8h*)(Arow + kb + half * 8);
        af.h[1] = *(const v8h*)(Arow + kb + 16 + half * 8);
#pragma unroll
        for (int t = 0; t < 4; ++t) {
            // B 32x16 f16 fragment: per-lane halves at K = kb+half*8..+7 and kb+16+half*8..+7
            const _Float16* Wr = Wrow0 + (size_t)(t * 16) * 128 + kb + half * 8;
            union { v16h v; v8h h[2]; } bf;
            bf.h[0] = *(const v8h*)(Wr);
            bf.h[1] = *(const v8h*)(Wr + 16);
            acc[t] = __builtin_amdgcn_wmma_f32_16x16x32_f16(false, af.v, false, bf.v,
                                                            (short)0, acc[t], false, false);
        }
    }

#pragma unroll
    for (int t = 0; t < 4; ++t) {
        int n = ng * 64 + t * 16 + rl;
        float bn = bias ? bias[n] : 0.0f;
#pragma unroll
        for (int v = 0; v < 8; ++v) {
            int m = mt * 16 + v + half * 8;      // C layout: VGPR v -> M=v (+8 upper lanes)
            C[(size_t)m * ldc + n] = acc[t][v] + bn;
        }
    }
}

// ---------------- LayerNorm + ReLU, in place (wave per row) ----------------
__global__ void k_ln_relu(float* __restrict__ x, int rowStride, int rows,
                          const float* __restrict__ g, const float* __restrict__ b) {
    int wave = (blockIdx.x * blockDim.x + threadIdx.x) >> 5;
    int lane = threadIdx.x & 31;
    if (wave >= rows) return;
    float* r = x + (size_t)wave * rowStride;
    float4 t = ((float4*)r)[lane];
    float mu = waveReduceSum(t.x + t.y + t.z + t.w) * (1.0f / 128.0f);
    float dx = t.x - mu, dy = t.y - mu, dz = t.z - mu, dw = t.w - mu;
    float var = waveReduceSum(dx * dx + dy * dy + dz * dz + dw * dw) * (1.0f / 128.0f);
    float inv = rsqrtf(var + 1e-5f);
    int d0 = lane * 4;
    t.x = fmaxf(dx * inv * g[d0 + 0] + b[d0 + 0], 0.0f);
    t.y = fmaxf(dy * inv * g[d0 + 1] + b[d0 + 1], 0.0f);
    t.z = fmaxf(dz * inv * g[d0 + 2] + b[d0 + 2], 0.0f);
    t.w = fmaxf(dw * inv * g[d0 + 3] + b[d0 + 3], 0.0f);
    ((float4*)r)[lane] = t;
}

// ---------------- attention over P=2 (wave per (item, head)); o overwrites q --------
__global__ void k_attn(float* __restrict__ qkv, int items) {
    int wave = (blockIdx.x * blockDim.x + threadIdx.x) >> 5;
    int lane = threadIdx.x & 31;
    int item = wave >> 2;
    int head = wave & 3;
    if (item >= items) return;
    float* r0 = qkv + (size_t)(item * 2 + 0) * 384;
    float* r1 = qkv + (size_t)(item * 2 + 1) * 384;
    int off = head * 32 + lane;                  // dh = 32, lane per element
    float q0 = r0[off],       q1 = r1[off];
    float k0 = r0[128 + off], k1 = r1[128 + off];
    float v0 = r0[256 + off], v1 = r1[256 + off];
    const float scale = 0.1767766952966369f;     // 1/sqrt(32)
    float s00 = waveReduceSum(q0 * k0) * scale;
    float s01 = waveReduceSum(q0 * k1) * scale;
    float s10 = waveReduceSum(q1 * k0) * scale;
    float s11 = waveReduceSum(q1 * k1) * scale;
    float m0 = fmaxf(s00, s01);
    float e00 = __expf(s00 - m0), e01 = __expf(s01 - m0);
    float z0 = 1.0f / (e00 + e01);
    float m1 = fmaxf(s10, s11);
    float e10 = __expf(s10 - m1), e11 = __expf(s11 - m1);
    float z1 = 1.0f / (e10 + e11);
    r0[off] = (e00 * v0 + e01 * v1) * z0;        // q region reused as o
    r1[off] = (e10 * v0 + e11 * v1) * z1;
}

// ---------------- residual + LN + mean over P (2 waves per item) ----------------
__global__ void k_final(const float* __restrict__ attn, const float* __restrict__ x,
                        const float* __restrict__ g, const float* __restrict__ b,
                        float* __restrict__ out, int items) {
    __shared__ float lnv[2][128];
    int item = blockIdx.x;
    int p    = threadIdx.x >> 5;
    int lane = threadIdx.x & 31;
    const float4* ar = (const float4*)(attn + (size_t)(item * 2 + p) * 128);
    const float4* xr = (const float4*)(x    + (size_t)(item * 2 + p) * 128);
    float4 t = ar[lane], u = xr[lane];
    t.x += u.x; t.y += u.y; t.z += u.z; t.w += u.w;
    float mu = waveReduceSum(t.x + t.y + t.z + t.w) * (1.0f / 128.0f);
    float dx = t.x - mu, dy = t.y - mu, dz = t.z - mu, dw = t.w - mu;
    float var = waveReduceSum(dx * dx + dy * dy + dz * dz + dw * dw) * (1.0f / 128.0f);
    float inv = rsqrtf(var + 1e-5f);
    int d0 = lane * 4;
    lnv[p][d0 + 0] = dx * inv * g[d0 + 0] + b[d0 + 0];
    lnv[p][d0 + 1] = dy * inv * g[d0 + 1] + b[d0 + 1];
    lnv[p][d0 + 2] = dz * inv * g[d0 + 2] + b[d0 + 2];
    lnv[p][d0 + 3] = dw * inv * g[d0 + 3] + b[d0 + 3];
    __syncthreads();
    float* o = out + (size_t)item * 128;
    for (int i = threadIdx.x; i < 128; i += 64)
        o[i] = 0.5f * (lnv[0][i] + lnv[1][i]);
}

extern "C" void kernel_launch(void* const* d_in, const int* in_sizes, int n_in,
                              void* d_out, int out_size, void* d_ws, size_t ws_size,
                              hipStream_t stream) {
    const float* feat_A = (const float*)d_in[0];
    const float* feat_B = (const float*)d_in[1];
    const int*   src_ab = (const int*)d_in[2];
    const int*   dst_ab = (const int*)d_in[3];
    const float* val_ab = (const float*)d_in[4];
    const int*   src_ba = (const int*)d_in[5];
    const int*   dst_ba = (const int*)d_in[6];
    const float* val_ba = (const float*)d_in[7];
    const float* W1  = (const float*)d_in[8];
    const float* b1  = (const float*)d_in[9];
    const float* g1  = (const float*)d_in[10];
    const float* be1 = (const float*)d_in[11];
    const float* W2  = (const float*)d_in[12];
    const float* b2  = (const float*)d_in[13];
    const float* g2  = (const float*)d_in[14];
    const float* be2 = (const float*)d_in[15];
    const float* Win  = (const float*)d_in[16];
    const float* bin_ = (const float*)d_in[17];
    const float* Wout = (const float*)d_in[18];
    const float* bout = (const float*)d_in[19];
    const float* lng  = (const float*)d_in[20];
    const float* lnb  = (const float*)d_in[21];
    float* out = (float*)d_out;

    float* ws = (float*)d_ws;
    const size_t NBV = (size_t)NB * VV * DD;       // 7,680,000 floats
    const size_t NAV = (size_t)NA * VV * DD;       // 7,680,000 floats
    float* hB   = ws;                              // [NB*V, 128]
    float* hA   = hB  + NBV;                       // [NA*V, 128]  (reused twice)
    float* xbuf = hA  + NAV;                       // [NA*V, 2, 128]
    float* qkv  = xbuf + NAV * 2;                  // [NA*V*2, 384]
    float* attn = qkv + (size_t)NA * VV * 2 * 384; // [NA*V*2, 128]
    _Float16* a16 = (_Float16*)(attn + NAV * 2);   // [NA*V*2, 128] f16 staging
    _Float16* w1h = a16 + NAV * 2;                 // 128*128 halfs
    _Float16* w2h = w1h + 128 * 128;
    _Float16* wih = w2h + 128 * 128;               // 384*128 halfs
    _Float16* woh = wih + 384 * 128;

    const int rowsB = NB * VV;                     // 60000
    const int rowsA = NA * VV;                     // 60000
    const int rowsX = NA * VV * 2;                 // 120000
    const int n4    = (int)(NBV / 4);

    dim3 blk(256);

    // ---- one-time weight conversions (f32 -> f16) ----
    k_cvt_f16<<<dim3(128 / 8), blk, 0, stream>>>(W1,   128, w1h, 128);
    k_cvt_f16<<<dim3(128 / 8), blk, 0, stream>>>(W2,   128, w2h, 128);
    k_cvt_f16<<<dim3(384 / 8), blk, 0, stream>>>(Win,  128, wih, 384);
    k_cvt_f16<<<dim3(128 / 8), blk, 0, stream>>>(Wout, 128, woh, 128);

    // ---- metapath A->B->A ----
    k_zero<<<dim3(n4 / 256), blk, 0, stream>>>(hB, n4);
    k_spmm<<<dim3((NEDGE * 64) / 256), blk, 0, stream>>>(feat_A, src_ab, dst_ab, val_ab, hB, NEDGE);
    k_l2norm<<<dim3(rowsB / 8), blk, 0, stream>>>(hB, rowsB);
    k_zero<<<dim3(n4 / 256), blk, 0, stream>>>(hA, n4);
    k_spmm<<<dim3((NEDGE * 64) / 256), blk, 0, stream>>>(hB, src_ba, dst_ba, val_ba, hA, NEDGE);
    k_l2norm<<<dim3(rowsA / 8), blk, 0, stream>>>(hA, rowsA);
    // h1 = relu(LN(hA @ W1^T + b1)) -> x[:, :, 0, :]
    k_cvt_f16<<<dim3(rowsA / 8), blk, 0, stream>>>(hA, 128, a16, rowsA);
    k_gemm_wmma<<<dim3(((rowsA / 16) * 2 + 7) / 8), blk, 0, stream>>>(
        a16, w1h, b1, xbuf, 256, rowsA / 16, 2);
    k_ln_relu<<<dim3(rowsA / 8), blk, 0, stream>>>(xbuf, 256, rowsA, g1, be1);

    // ---- metapath B->A ----
    k_zero<<<dim3(n4 / 256), blk, 0, stream>>>(hA, n4);
    k_spmm<<<dim3((NEDGE * 64) / 256), blk, 0, stream>>>(feat_B, src_ba, dst_ba, val_ba, hA, NEDGE);
    k_l2norm<<<dim3(rowsA / 8), blk, 0, stream>>>(hA, rowsA);
    // h2 = relu(LN(hA @ W2^T + b2)) -> x[:, :, 1, :]
    k_cvt_f16<<<dim3(rowsA / 8), blk, 0, stream>>>(hA, 128, a16, rowsA);
    k_gemm_wmma<<<dim3(((rowsA / 16) * 2 + 7) / 8), blk, 0, stream>>>(
        a16, w2h, b2, xbuf + 128, 256, rowsA / 16, 2);
    k_ln_relu<<<dim3(rowsA / 8), blk, 0, stream>>>(xbuf + 128, 256, rowsA, g2, be2);

    // ---- attention over P ----
    // qkv = x @ Win^T + bin   : [120000, 384]
    k_cvt_f16<<<dim3(rowsX / 8), blk, 0, stream>>>(xbuf, 128, a16, rowsX);
    k_gemm_wmma<<<dim3(((rowsX / 16) * 6 + 7) / 8), blk, 0, stream>>>(
        a16, wih, bin_, qkv, 384, rowsX / 16, 6);
    // per-(item, head) softmax attention; o written into q region of qkv
    k_attn<<<dim3((rowsA * HEADS) / 8), blk, 0, stream>>>(qkv, rowsA);
    // attn = o @ Wout^T + bout : [120000, 128]  (o rows strided by 384)
    k_cvt_f16<<<dim3(rowsX / 8), blk, 0, stream>>>(qkv, 384, a16, rowsX);
    k_gemm_wmma<<<dim3(((rowsX / 16) * 2 + 7) / 8), blk, 0, stream>>>(
        a16, woh, bout, attn, 128, rowsX / 16, 2);

    // ---- residual + LN + mean over P ----
    k_final<<<dim3(rowsA), dim3(64), 0, stream>>>(attn, xbuf, lng, lnb, out, rowsA);
}
